// LMultiHeadAttention_10110353014884
// MI455X (gfx1250) — matrix operand order repaired and checked by reading, not compiled, not run on hardware
//
#include <hip/hip_runtime.h>

// MHA: S=2048, B=4, D=1024, H=16, DK=64.  f16 WMMA with f32 accumulation,
// flash-attention (no materialized score matrix), intermediates live in d_ws.
// Round 3: attention K/V tiles cooperatively staged into LDS with
// global_load_async_to_lds_b128 (ASYNCcnt, double-buffered) -- the 8 waves of
// a block share one (b,h) stream, so this cuts K/V fragment traffic 8x.

#define S_LEN 2048
#define BATCH 4
#define DMODEL 1024
#define NHEAD 16
#define DHEAD 64

typedef __attribute__((ext_vector_type(16))) _Float16 v16h;
typedef __attribute__((ext_vector_type(8)))  _Float16 v8h;
typedef __attribute__((ext_vector_type(8)))  float    v8f;
typedef __attribute__((ext_vector_type(4)))  float    v4f;

__device__ __forceinline__ v8f wmma16x16x32(v16h a, v16h b, v8f c) {
  // D = A(16x32 f16) x B(32x16 f16) + C(16x16 f32)
  return __builtin_amdgcn_wmma_f32_16x16x32_f16(false, a, false, b, (short)0, c,
                                                false, false);
}

// A-fragment (16x32 f16): lanes<16 hold K 0..15,
// lanes>=16 hold K 16..31; row M = lane%16.
__device__ __forceinline__ v16h load_a_f16(const _Float16* p, int lane) {
  return *(const v16h*)(p + ((lane & 16) ? 16 : 0));
}

__device__ __forceinline__ v16h load_a_f32(const float* p, int lane) {
  const float* p0 = p + ((lane & 16) ? 16 : 0);
  v4f x0 = *(const v4f*)(p0);
  v4f x1 = *(const v4f*)(p0 + 4);
  v4f x2 = *(const v4f*)(p0 + 8);
  v4f x3 = *(const v4f*)(p0 + 12);
  v16h a;
#pragma unroll
  for (int i = 0; i < 4; ++i) {
    a[i]      = (_Float16)x0[i];
    a[4 + i]  = (_Float16)x1[i];
    a[8 + i]  = (_Float16)x2[i];
    a[12 + i] = (_Float16)x3[i];
  }
  return a;
}

// B-fragment (32x16 f16).  Same k order as A: lanes<16 hold K 0..15, lanes>=16 hold
// K 16..31; col N = lane%16.  16 contiguous halves per lane.
__device__ __forceinline__ v16h load_b_f16(const _Float16* p, int lane) {
  return *(const v16h*)(p + ((lane & 16) ? 16 : 0));
}

// Async global -> LDS copy of 16 bytes per lane (VGPR-bypassing, ASYNCcnt).
__device__ __forceinline__ void async_copy_b128(const _Float16* gsrc,
                                                _Float16* ldst) {
  // Generic pointer to a __shared__ object: low 32 bits are the LDS offset.
  unsigned lds = (unsigned)(size_t)ldst;
  unsigned long long ga = (unsigned long long)(size_t)gsrc;
  asm volatile("global_load_async_to_lds_b128 %0, %1, off"
               :: "v"(lds), "v"(ga) : "memory");
}

__device__ __forceinline__ void wait_asynccnt_le2() {
  asm volatile("s_wait_asynccnt 0x2" ::: "memory");
}

// ---------------------------------------------------------------------------
// Kernel 0: one-shot fp32 -> f16 conversion (weights).  8 elements/thread.
// ---------------------------------------------------------------------------
__global__ __launch_bounds__(256) void cvt_f32_to_f16(
    const float* __restrict__ src, _Float16* __restrict__ dst, int n8) {
  const int i = blockIdx.x * 256 + threadIdx.x;
  if (i >= n8) return;
  const v4f* s = (const v4f*)src + (size_t)i * 2;
  v4f a = s[0], b = s[1];
  v8h o;
#pragma unroll
  for (int j = 0; j < 4; ++j) {
    o[j] = (_Float16)a[j];
    o[4 + j] = (_Float16)b[j];
  }
  *((v8h*)dst + i) = o;
}

// ---------------------------------------------------------------------------
// Kernel 1: Q/K/V projections.  One wave computes a 16(M)x64(N) tile.
// Q,K -> [B,H,S,DK] f16;  V -> [B,H,DK,S] f16 (transposed for PV B-frags).
// Weights already f16: inner loop = b128 loads + x f32->f16 cvt + v_wmma.
// ---------------------------------------------------------------------------
__global__ __launch_bounds__(256) void qkv_proj(
    const float* __restrict__ xq, const float* __restrict__ xk,
    const float* __restrict__ xv, const _Float16* __restrict__ wq16,
    const _Float16* __restrict__ wk16, const _Float16* __restrict__ wv16,
    const float* __restrict__ bq, const float* __restrict__ bk,
    const float* __restrict__ bv, _Float16* __restrict__ q16,
    _Float16* __restrict__ k16, _Float16* __restrict__ vt16) {
  const int lane = threadIdx.x & 31;
  const int warp = threadIdx.x >> 5;
  const int flat = blockIdx.x * 8 + warp;   // 24576 waves total
  const int mat  = flat >> 13;              // 0:Q 1:K 2:V  (8192 waves each)
  const int rem  = flat & 8191;
  const int mt   = rem >> 4;                // 512 M-tiles (rows of [S*B, D])
  const int nt   = rem & 15;                // 16 N-tiles of 64
  const int m16  = lane & 15;

  const float* x       = (mat == 0) ? xq : ((mat == 1) ? xk : xv);
  const _Float16* W    = (mat == 0) ? wq16 : ((mat == 1) ? wk16 : wv16);
  const float* bias    = (mat == 0) ? bq : ((mat == 1) ? bk : bv);

  const float* xrow      = x + (size_t)(mt * 16 + m16) * DMODEL;  // A row m
  const _Float16* wbase  = W + (size_t)(nt * 64 + m16) * DMODEL;  // B col n

  v8f acc[4] = {};
  for (int kc = 0; kc < DMODEL; kc += 32) {
    v16h a = load_a_f32(xrow + kc, lane);
#pragma unroll
    for (int t = 0; t < 4; ++t) {
      v16h bf = load_b_f16(wbase + (size_t)(t * 16) * DMODEL + kc, lane);
      acc[t] = wmma16x16x32(a, bf, acc[t]);
    }
  }

  const int rbase = (lane & 16) ? 8 : 0;
#pragma unroll
  for (int t = 0; t < 4; ++t) {
    const int n = nt * 64 + t * 16 + m16;
    const float bval = bias[n];
    const int hh = n >> 6;
    const int dk = n & 63;
#pragma unroll
    for (int r = 0; r < 8; ++r) {
      const int m  = mt * 16 + rbase + r;  // row of [S,B,D]: m = s*B + b
      const int ss = m >> 2;
      const int bb = m & 3;
      const float val = acc[t][r] + bval;
      if (mat == 2) {
        vt16[(((size_t)bb * NHEAD + hh) * DHEAD + dk) * S_LEN + ss] =
            (_Float16)val;
      } else {
        _Float16* dst = (mat == 0) ? q16 : k16;
        dst[(((size_t)bb * NHEAD + hh) * S_LEN + ss) * DHEAD + dk] =
            (_Float16)val;
      }
    }
  }
}

// ---------------------------------------------------------------------------
// Kernel 2: flash attention.  8 waves/block share one (b,h); per 32-key block
// the K tile (32x64 f16) and V tile (64x32 f16) are staged into LDS by
// global_load_async_to_lds_b128 (double-buffered, ASYNCcnt + barriers).
// Each wave: S=QK^T (4 WMMA), online softmax via shfl_xor, P transposed
// through per-wave LDS staging, O += P@V (4 WMMA).
// ---------------------------------------------------------------------------
__global__ __launch_bounds__(256) void attn_flash(
    const _Float16* __restrict__ q16, const _Float16* __restrict__ k16,
    const _Float16* __restrict__ vt16, _Float16* __restrict__ o16) {
  __shared__ _Float16 pstage[8 * 16 * 32];   // 8 KB: per-wave P transpose
  __shared__ _Float16 ktile[2][32 * 64];     // 2 x 4 KB, [s][dk]
  __shared__ _Float16 vtile[2][64 * 32];     // 2 x 4 KB, [dk][s]

  const int tid  = threadIdx.x;
  const int lane = tid & 31;
  const int warp = tid >> 5;
  const int flat = blockIdx.x * 8 + warp;   // 8192 waves total
  const int qt = flat & 127;                // S/16 query tiles
  const int h  = (flat >> 7) & 15;          // uniform across block (8 | 128)
  const int b  = flat >> 11;
  const int m16  = lane & 15;
  const int bcol = (lane & 16) ? 16 : 0;    // B-frag K chunk
  const int acol = (lane & 16) ? 16 : 0;    // A-frag K chunk
  const int rbase = (lane & 16) ? 8 : 0;    // C-frag row base

  const _Float16* qbase =
      q16 + (((size_t)(b * NHEAD + h)) * S_LEN + (size_t)qt * 16) * DHEAD;
  const _Float16* kbase = k16 + ((size_t)(b * NHEAD + h)) * S_LEN * DHEAD;
  const _Float16* vbase = vt16 + ((size_t)(b * NHEAD + h)) * DHEAD * S_LEN;

  // Per-thread async-copy slices: K tile = 256 x 16B (8 chunks/row of 64h),
  // V tile = 256 x 16B (4 chunks/row of 32h).
  const int krow = tid >> 3, kchk = (tid & 7) * 8;
  const int vrow = tid >> 2, vchk = (tid & 3) * 8;

  // Q A-fragments: 16 rows x DK=64 = two 16x32 frags, held for whole loop.
  v16h qa0 = load_a_f16(qbase + (size_t)m16 * DHEAD, lane);
  v16h qa1 = load_a_f16(qbase + (size_t)m16 * DHEAD + 32, lane);

  v8f oacc[4] = {};
  float mrow[8], lrow[8];
#pragma unroll
  for (int r = 0; r < 8; ++r) { mrow[r] = -1e30f; lrow[r] = 0.0f; }

  _Float16* pbuf = pstage + warp * 512;

  // Prologue: stage block 0 into buffer 0.
  async_copy_b128(kbase + (size_t)krow * DHEAD + kchk, &ktile[0][krow * 64 + kchk]);
  async_copy_b128(vbase + (size_t)vrow * S_LEN + vchk, &vtile[0][vrow * 32 + vchk]);

  for (int kb = 0; kb < S_LEN / 32; ++kb) {
    const int s0  = kb * 32;
    const int cur = kb & 1;
    const int nxt = cur ^ 1;

    // Stage block kb+1 (last iteration overshoots into adjacent ws: harmless,
    // that buffer is never consumed).
    const int s1 = s0 + 32;
    async_copy_b128(kbase + (size_t)(s1 + krow) * DHEAD + kchk,
                    &ktile[nxt][krow * 64 + kchk]);
    async_copy_b128(vbase + (size_t)vrow * S_LEN + s1 + vchk,
                    &vtile[nxt][vrow * 32 + vchk]);

    // Retire this buffer's copies (2 newer ones may stay in flight), then
    // make them visible block-wide.
    wait_asynccnt_le2();
    __syncthreads();

    const _Float16* kt = &ktile[cur][0];
    const _Float16* vt = &vtile[cur][0];

    // K B-fragments from LDS: B[k=dk][n=s_k] = K[s_k][dk].
    v16h k00 = *(const v16h*)(kt + (m16)*64 + bcol);
    v16h k01 = *(const v16h*)(kt + (m16)*64 + 32 + bcol);
    v16h k10 = *(const v16h*)(kt + (16 + m16) * 64 + bcol);
    v16h k11 = *(const v16h*)(kt + (16 + m16) * 64 + 32 + bcol);

    v8f sc0 = {}, sc1 = {};
    sc0 = wmma16x16x32(qa0, k00, sc0);
    sc0 = wmma16x16x32(qa1, k01, sc0);
    sc1 = wmma16x16x32(qa0, k10, sc1);
    sc1 = wmma16x16x32(qa1, k11, sc1);

    // Online softmax over the 32 keys.  C-layout row r lives in 16-lane
    // half-groups, so xor-shuffles 1/2/4/8 reduce across the row.
    float corr[8];
#pragma unroll
    for (int r = 0; r < 8; ++r) {
      float a = sc0[r] * 0.125f;  // 1/sqrt(DK)
      float c = sc1[r] * 0.125f;
      float mx = fmaxf(a, c);
      mx = fmaxf(mx, __shfl_xor(mx, 1));
      mx = fmaxf(mx, __shfl_xor(mx, 2));
      mx = fmaxf(mx, __shfl_xor(mx, 4));
      mx = fmaxf(mx, __shfl_xor(mx, 8));
      float mn = fmaxf(mrow[r], mx);
      float cr = __expf(mrow[r] - mn);
      float p0 = __expf(a - mn);
      float p1 = __expf(c - mn);
      float rs = p0 + p1;
      rs += __shfl_xor(rs, 1);
      rs += __shfl_xor(rs, 2);
      rs += __shfl_xor(rs, 4);
      rs += __shfl_xor(rs, 8);
      lrow[r] = lrow[r] * cr + rs;
      mrow[r] = mn;
      corr[r] = cr;
      sc0[r] = p0;
      sc1[r] = p1;
    }
#pragma unroll
    for (int t = 0; t < 4; ++t)
#pragma unroll
      for (int r = 0; r < 8; ++r) oacc[t][r] *= corr[r];

    // P: C-layout -> LDS [16 rows][32 cols] -> reload as A-fragment.
#pragma unroll
    for (int r = 0; r < 8; ++r) {
      pbuf[(rbase + r) * 32 + m16]      = (_Float16)sc0[r];
      pbuf[(rbase + r) * 32 + 16 + m16] = (_Float16)sc1[r];
    }
    v16h pa;
    {
      const _Float16* pr = pbuf + m16 * 32;
      v8h x0 = *(const v8h*)(pr + acol);
      v8h x1 = *(const v8h*)(pr + acol + 8);
#pragma unroll
      for (int i = 0; i < 8; ++i) { pa[i] = x0[i]; pa[8 + i] = x1[i]; }
    }

    // O += P(16x32) @ V(32x64): V tile is [dk][s] so B-frags are contiguous.
#pragma unroll
    for (int t = 0; t < 4; ++t) {
      v16h vb = *(const v16h*)(vt + (t * 16 + m16) * 32 + bcol);
      oacc[t] = wmma16x16x32(pa, vb, oacc[t]);
    }

    // All waves done with buf[cur] before it is overwritten next iteration.
    __syncthreads();
  }

  // Epilogue: O /= l, store f16 into [B,S,H,DK] for the output GEMM.
#pragma unroll
  for (int r = 0; r < 8; ++r) {
    const float inv = 1.0f / lrow[r];
    const int s = qt * 16 + rbase + r;
#pragma unroll
    for (int t = 0; t < 4; ++t) {
      const int dk = t * 16 + m16;
      o16[(((size_t)b * S_LEN + s) * NHEAD + h) * DHEAD + dk] =
          (_Float16)(oacc[t][r] * inv);
    }
  }
}

// ---------------------------------------------------------------------------
// Kernel 3: output projection.  O[B*S, D] f16 x Wo^T(f16) + bo -> [S,B,D] f32.
// Fully f16 operands: inner loop is pure b128 loads + v_wmma.
// ---------------------------------------------------------------------------
__global__ __launch_bounds__(256) void out_proj(
    const _Float16* __restrict__ o16, const _Float16* __restrict__ wo16,
    const float* __restrict__ bo, float* __restrict__ out) {
  const int lane = threadIdx.x & 31;
  const int warp = threadIdx.x >> 5;
  const int flat = blockIdx.x * 8 + warp;  // 8192 waves
  const int mt = flat >> 4;
  const int nt = flat & 15;
  const int m16 = lane & 15;

  const _Float16* arow  = o16 + (size_t)(mt * 16 + m16) * DMODEL;
  const _Float16* wbase = wo16 + (size_t)(nt * 64 + m16) * DMODEL;

  v8f acc[4] = {};
  for (int kc = 0; kc < DMODEL; kc += 32) {
    v16h a = load_a_f16(arow + kc, lane);
#pragma unroll
    for (int t = 0; t < 4; ++t) {
      v16h bf = load_b_f16(wbase + (size_t)(t * 16) * DMODEL + kc, lane);
      acc[t] = wmma16x16x32(a, bf, acc[t]);
    }
  }

  const int rbase = (lane & 16) ? 8 : 0;
#pragma unroll
  for (int t = 0; t < 4; ++t) {
    const int n = nt * 64 + t * 16 + m16;
    const float bval = bo[n];
#pragma unroll
    for (int r = 0; r < 8; ++r) {
      const int m  = mt * 16 + rbase + r;  // m = b*S + s (o16 row order)
      const int bb = m >> 11;
      const int ss = m & 2047;
      out[((size_t)ss * BATCH + bb) * DMODEL + n] = acc[t][r] + bval;
    }
  }
}

extern "C" void kernel_launch(void* const* d_in, const int* in_sizes, int n_in,
                              void* d_out, int out_size, void* d_ws,
                              size_t ws_size, hipStream_t stream) {
  (void)in_sizes; (void)n_in; (void)out_size; (void)ws_size;
  const float* xq = (const float*)d_in[0];
  const float* xk = (const float*)d_in[1];
  const float* xv = (const float*)d_in[2];
  const float* Wq = (const float*)d_in[3];
  const float* bq = (const float*)d_in[4];
  const float* Wk = (const float*)d_in[5];
  const float* bk = (const float*)d_in[6];
  const float* Wv = (const float*)d_in[7];
  const float* bv = (const float*)d_in[8];
  const float* Wo = (const float*)d_in[9];
  const float* bo = (const float*)d_in[10];
  float* out = (float*)d_out;

  // Workspace layout (halves):
  //   q16/k16/vt16/o16 : 4 x B*H*S*DK = 4 x 8,388,608  (16 MB each)
  //   wq/wk/wv/wo f16  : 4 x D*D      = 4 x 1,048,576  ( 2 MB each)
  // Total 72 MB.
  const size_t elems = (size_t)BATCH * NHEAD * S_LEN * DHEAD;
  const size_t welems = (size_t)DMODEL * DMODEL;
  _Float16* q16  = (_Float16*)d_ws;
  _Float16* k16  = q16 + elems;
  _Float16* vt16 = k16 + elems;
  _Float16* o16  = vt16 + elems;
  _Float16* wq16 = o16 + elems;
  _Float16* wk16 = wq16 + welems;
  _Float16* wv16 = wk16 + welems;
  _Float16* wo16 = wv16 + welems;

  // One-shot weight conversion: D*D/8 threads each.
  const int n8 = (int)(welems / 8);            // 131072
  const int cblk = (n8 + 255) / 256;           // 512 blocks
  cvt_f32_to_f16<<<cblk, 256, 0, stream>>>(Wq, wq16, n8);
  cvt_f32_to_f16<<<cblk, 256, 0, stream>>>(Wk, wk16, n8);
  cvt_f32_to_f16<<<cblk, 256, 0, stream>>>(Wv, wv16, n8);
  cvt_f32_to_f16<<<cblk, 256, 0, stream>>>(Wo, wo16, n8);

  // 3*512*16 output tiles, 8 waves (256 thr) per block.
  qkv_proj<<<3072, 256, 0, stream>>>(xq, xk, xv, wq16, wk16, wv16, bq, bk, bv,
                                     q16, k16, vt16);
  // B*H*(S/16) = 8192 q-tile waves.
  attn_flash<<<1024, 256, 0, stream>>>(q16, k16, vt16, o16);
  // 512*16 output tiles.
  out_proj<<<1024, 256, 0, stream>>>(o16, wo16, bo, out);
}
